// SingleHeadSelfAttention_19061064860283
// MI455X (gfx1250) — matrix operand order repaired and checked by reading, not compiled, not run on hardware
//
#include <hip/hip_runtime.h>
#include <hip/hip_bf16.h>
#include <math.h>

// ---- CDNA5 WMMA types ----
typedef __bf16 bf16_t;
typedef bf16_t v4bf  __attribute__((ext_vector_type(4)));
typedef bf16_t v8bf  __attribute__((ext_vector_type(8)));
typedef bf16_t v16bf __attribute__((ext_vector_type(16)));
typedef float  v8f   __attribute__((ext_vector_type(8)));

#define N_DIM   768
#define SEQ     2048
#define BATCH   8
#define QK_LD   1536                  // q|k packed row stride
#define ROWS    (BATCH * SEQ)         // 16384
#define SCALE   0.03608439182435161f  // 1/sqrt(768)

// ---------------------------------------------------------------------------
// Operand loads (all contiguous, 16B/32B aligned):
// A 16x32 bf16: lane=M%16, half=lane/16;
//   K=16*half+j  => one 16-elem run (matches B).
__device__ __forceinline__ v16bf load_a_bf(const bf16_t* __restrict__ base,
                                           int lda, int lane) {
    int row = lane & 15, half = lane >> 4;
    return *(const v16bf*)(base + (size_t)row * lda + 16 * half);
}

// B 32x16 bf16: N=lane%16, K=j+16*(lane/16). Source is "B^T" row-major
// [N x K] with stride ld -> 16 contiguous elements per lane (32B load).
__device__ __forceinline__ v16bf load_bt_bf(const bf16_t* __restrict__ base,
                                            int ld, int lane) {
    int n = lane & 15, half = lane >> 4;
    return *(const v16bf*)(base + (size_t)n * ld + 16 * half);
}

// Error-compensated accumulate: C += Ah*Bh + Ah*Bl + Al*Bh (fp32 acc).
__device__ __forceinline__ v8f wmma3(v16bf ah, v16bf al, v16bf bh, v16bf bl, v8f c) {
    c = __builtin_amdgcn_wmma_f32_16x16x32_bf16(false, ah, false, bh, (short)0, c, false, false);
    c = __builtin_amdgcn_wmma_f32_16x16x32_bf16(false, ah, false, bl, (short)0, c, false, false);
    c = __builtin_amdgcn_wmma_f32_16x16x32_bf16(false, al, false, bh, (short)0, c, false, false);
    return c;
}

// C/D 16x16 f32 layout: VGPR r -> row = r + 8*(lane/16), col = lane%16.
__device__ __forceinline__ void store_c_f32(float* __restrict__ C, int ldc,
                                            int lane, v8f c) {
    int n = lane & 15, half = lane >> 4;
#pragma unroll
    for (int r = 0; r < 8; ++r)
        C[(size_t)(r + 8 * half) * ldc + n] = c[r];
}

// Store tile split into bf16 hi/lo arrays (row-major).
__device__ __forceinline__ void store_c_split(bf16_t* __restrict__ Hi,
                                              bf16_t* __restrict__ Lo,
                                              int ldc, int lane, v8f c, float scale) {
    int n = lane & 15, half = lane >> 4;
#pragma unroll
    for (int r = 0; r < 8; ++r) {
        float x = c[r] * scale;
        bf16_t h = (bf16_t)x;
        size_t off = (size_t)(r + 8 * half) * ldc + n;
        Hi[off] = h;
        Lo[off] = (bf16_t)(x - (float)h);
    }
}

// Store tile TRANSPOSED into hi/lo arrays: out[col][row]; rows (r) become the
// contiguous dim -> one packed 16B store per array per lane.
__device__ __forceinline__ void store_c_split_T(bf16_t* __restrict__ Hi,
                                                bf16_t* __restrict__ Lo,
                                                int ldt, int lane, v8f c) {
    int n = lane & 15, half = lane >> 4;
    v8bf h, l;
#pragma unroll
    for (int r = 0; r < 8; ++r) {
        float x = c[r];
        bf16_t hh = (bf16_t)x;
        h[r] = hh;
        l[r] = (bf16_t)(x - (float)hh);
    }
    size_t off = (size_t)n * ldt + 8 * half;
    *(v8bf*)(Hi + off) = h;
    *(v8bf*)(Lo + off) = l;
}

// ---------------- split X -> Xh/Xl (row-major [16384,768]) ------------------
__global__ void split_x_kernel(const float* __restrict__ X,
                               bf16_t* __restrict__ Xh, bf16_t* __restrict__ Xl) {
    size_t i = ((size_t)blockIdx.x * 256 + threadIdx.x) * 4;
    float4 v = *(const float4*)(X + i);
    v4bf h, l;
    h[0] = (bf16_t)v.x; l[0] = (bf16_t)(v.x - (float)h[0]);
    h[1] = (bf16_t)v.y; l[1] = (bf16_t)(v.y - (float)h[1]);
    h[2] = (bf16_t)v.z; l[2] = (bf16_t)(v.z - (float)h[2]);
    h[3] = (bf16_t)v.w; l[3] = (bf16_t)(v.w - (float)h[3]);
    *(v4bf*)(Xh + i) = h;
    *(v4bf*)(Xl + i) = l;
}

// ------------- split + transpose W[768,2304] -> WT[2304,768] ----------------
__global__ void split_wT_kernel(const float* __restrict__ W,
                                bf16_t* __restrict__ WTh, bf16_t* __restrict__ WTl) {
    size_t id = (size_t)blockIdx.x * 256 + threadIdx.x;  // 2304*768
    size_t n = id / N_DIM, k = id % N_DIM;
    float x = W[k * (3 * N_DIM) + n];
    bf16_t h = (bf16_t)x;
    WTh[id] = h;
    WTl[id] = (bf16_t)(x - (float)h);
}

// ---------------- Kernel 1: QKV = X @ W (2x2 tiles per wave) ----------------
// q,k stored row-major split [16384,1536]; v stored transposed split [8,768,2048].
__global__ void qkv_gemm_kernel(const bf16_t* __restrict__ Xh, const bf16_t* __restrict__ Xl,
                                const bf16_t* __restrict__ WTh, const bf16_t* __restrict__ WTl,
                                bf16_t* __restrict__ QKh, bf16_t* __restrict__ QKl,
                                bf16_t* __restrict__ Vth, bf16_t* __restrict__ Vtl) {
    int lane = threadIdx.x & 31, wave = threadIdx.x >> 5;
    int m0 = blockIdx.x * 32;                  // 0..16352
    int n0 = (blockIdx.y * 8 + wave) * 32;     // 0..2272

    const bf16_t* A0h = Xh + (size_t)m0 * N_DIM;
    const bf16_t* A0l = Xl + (size_t)m0 * N_DIM;
    const bf16_t* B0h = WTh + (size_t)n0 * N_DIM;  // WT row n = output col n
    const bf16_t* B0l = WTl + (size_t)n0 * N_DIM;

    v8f c00 = {}, c01 = {}, c10 = {}, c11 = {};
#pragma unroll 2
    for (int k = 0; k < N_DIM; k += 32) {
        v16bf a0h = load_a_bf(A0h + k, N_DIM, lane);
        v16bf a0l = load_a_bf(A0l + k, N_DIM, lane);
        v16bf a1h = load_a_bf(A0h + (size_t)16 * N_DIM + k, N_DIM, lane);
        v16bf a1l = load_a_bf(A0l + (size_t)16 * N_DIM + k, N_DIM, lane);
        v16bf b0h = load_bt_bf(B0h + k, N_DIM, lane);
        v16bf b0l = load_bt_bf(B0l + k, N_DIM, lane);
        v16bf b1h = load_bt_bf(B0h + (size_t)16 * N_DIM + k, N_DIM, lane);
        v16bf b1l = load_bt_bf(B0l + (size_t)16 * N_DIM + k, N_DIM, lane);
        c00 = wmma3(a0h, a0l, b0h, b0l, c00);
        c01 = wmma3(a0h, a0l, b1h, b1l, c01);
        c10 = wmma3(a1h, a1l, b0h, b0l, c10);
        c11 = wmma3(a1h, a1l, b1h, b1l, c11);
    }

    if (n0 < QK_LD) {        // q|k region (boundary 1536 is a multiple of 32)
        bf16_t* H = QKh + (size_t)m0 * QK_LD + n0;
        bf16_t* L = QKl + (size_t)m0 * QK_LD + n0;
        store_c_split(H, L, QK_LD, lane, c00, 1.f);
        store_c_split(H + 16, L + 16, QK_LD, lane, c01, 1.f);
        store_c_split(H + (size_t)16 * QK_LD, L + (size_t)16 * QK_LD, QK_LD, lane, c10, 1.f);
        store_c_split(H + (size_t)16 * QK_LD + 16, L + (size_t)16 * QK_LD + 16, QK_LD, lane, c11, 1.f);
    } else {                 // v region -> transposed [b][feature][seq]
        int b  = m0 / SEQ;
        int s0 = m0 - b * SEQ;
        int f0 = n0 - QK_LD;
        bf16_t* H = Vth + ((size_t)b * N_DIM + f0) * SEQ + s0;
        bf16_t* L = Vtl + ((size_t)b * N_DIM + f0) * SEQ + s0;
        store_c_split_T(H, L, SEQ, lane, c00);                                   // (s0,   f0)
        store_c_split_T(H + (size_t)16 * SEQ, L + (size_t)16 * SEQ, SEQ, lane, c01); // (s0,   f0+16)
        store_c_split_T(H + 16, L + 16, SEQ, lane, c10);                         // (s0+16,f0)
        store_c_split_T(H + (size_t)16 * SEQ + 16, L + (size_t)16 * SEQ + 16, SEQ, lane, c11);
    }
}

// ------------- Kernel 2: S[b] = (Q @ K^T)*scale, stored split ---------------
__global__ void scores_gemm_kernel(const bf16_t* __restrict__ QKh, const bf16_t* __restrict__ QKl,
                                   bf16_t* __restrict__ Sh, bf16_t* __restrict__ Sl) {
    int lane = threadIdx.x & 31, wave = threadIdx.x >> 5;
    int b  = blockIdx.z;
    int m0 = blockIdx.x * 32;                  // query rows
    int n0 = (blockIdx.y * 8 + wave) * 32;     // key rows

    const bf16_t* Qh = QKh + ((size_t)b * SEQ + m0) * QK_LD;          // cols [0,768)
    const bf16_t* Ql = QKl + ((size_t)b * SEQ + m0) * QK_LD;
    const bf16_t* Kh = QKh + ((size_t)b * SEQ + n0) * QK_LD + N_DIM;  // cols [768,1536)
    const bf16_t* Kl = QKl + ((size_t)b * SEQ + n0) * QK_LD + N_DIM;

    v8f c00 = {}, c01 = {}, c10 = {}, c11 = {};
#pragma unroll 2
    for (int k = 0; k < N_DIM; k += 32) {
        v16bf a0h = load_a_bf(Qh + k, QK_LD, lane);
        v16bf a0l = load_a_bf(Ql + k, QK_LD, lane);
        v16bf a1h = load_a_bf(Qh + (size_t)16 * QK_LD + k, QK_LD, lane);
        v16bf a1l = load_a_bf(Ql + (size_t)16 * QK_LD + k, QK_LD, lane);
        v16bf b0h = load_bt_bf(Kh + k, QK_LD, lane);
        v16bf b0l = load_bt_bf(Kl + k, QK_LD, lane);
        v16bf b1h = load_bt_bf(Kh + (size_t)16 * QK_LD + k, QK_LD, lane);
        v16bf b1l = load_bt_bf(Kl + (size_t)16 * QK_LD + k, QK_LD, lane);
        c00 = wmma3(a0h, a0l, b0h, b0l, c00);
        c01 = wmma3(a0h, a0l, b1h, b1l, c01);
        c10 = wmma3(a1h, a1l, b0h, b0l, c10);
        c11 = wmma3(a1h, a1l, b1h, b1l, c11);
    }

    bf16_t* H = Sh + ((size_t)b * SEQ + m0) * SEQ + n0;
    bf16_t* L = Sl + ((size_t)b * SEQ + m0) * SEQ + n0;
    store_c_split(H, L, SEQ, lane, c00, SCALE);
    store_c_split(H + 16, L + 16, SEQ, lane, c01, SCALE);
    store_c_split(H + (size_t)16 * SEQ, L + (size_t)16 * SEQ, SEQ, lane, c10, SCALE);
    store_c_split(H + (size_t)16 * SEQ + 16, L + (size_t)16 * SEQ + 16, SEQ, lane, c11, SCALE);
}

// -------- Kernel 3: row softmax over hi+lo, rewrite hi/lo in place ----------
__global__ void softmax_rows_kernel(bf16_t* __restrict__ Sh, bf16_t* __restrict__ Sl) {
    bf16_t* ph = Sh + (size_t)blockIdx.x * SEQ;
    bf16_t* pl = Sl + (size_t)blockIdx.x * SEQ;
    int t = threadIdx.x;                 // 256 threads, 8 elems each
    __shared__ float red[8];

    float vals[8];
    float lmax = -INFINITY;
#pragma unroll
    for (int i = 0; i < 8; ++i) {
        int idx = t + i * 256;
        vals[i] = (float)ph[idx] + (float)pl[idx];
        lmax = fmaxf(lmax, vals[i]);
    }
#pragma unroll
    for (int o = 16; o > 0; o >>= 1) lmax = fmaxf(lmax, __shfl_xor(lmax, o, 32));
    if ((t & 31) == 0) red[t >> 5] = lmax;
    __syncthreads();
    float m = red[0];
#pragma unroll
    for (int w = 1; w < 8; ++w) m = fmaxf(m, red[w]);
    __syncthreads();

    float lsum = 0.f;
#pragma unroll
    for (int i = 0; i < 8; ++i) {
        vals[i] = __expf(vals[i] - m);
        lsum += vals[i];
    }
#pragma unroll
    for (int o = 16; o > 0; o >>= 1) lsum += __shfl_xor(lsum, o, 32);
    if ((t & 31) == 0) red[t >> 5] = lsum;
    __syncthreads();
    float s = 0.f;
#pragma unroll
    for (int w = 0; w < 8; ++w) s += red[w];
    float inv = 1.0f / s;
#pragma unroll
    for (int i = 0; i < 8; ++i) {
        int idx = t + i * 256;
        float y = vals[i] * inv;
        bf16_t h = (bf16_t)y;
        ph[idx] = h;
        pl[idx] = (bf16_t)(y - (float)h);
    }
}

// ---------------- Kernel 4: out[b] = P @ V (2x2 tiles per wave) -------------
__global__ void pv_gemm_kernel(const bf16_t* __restrict__ Sh, const bf16_t* __restrict__ Sl,
                               const bf16_t* __restrict__ Vth, const bf16_t* __restrict__ Vtl,
                               float* __restrict__ O) {
    int lane = threadIdx.x & 31, wave = threadIdx.x >> 5;
    int b  = blockIdx.z;
    int m0 = blockIdx.x * 32;                  // query rows
    int n0 = (blockIdx.y * 8 + wave) * 32;     // feature cols (0..767)

    const bf16_t* Ph = Sh + ((size_t)b * SEQ + m0) * SEQ;
    const bf16_t* Pl = Sl + ((size_t)b * SEQ + m0) * SEQ;
    const bf16_t* Bh = Vth + ((size_t)b * N_DIM + n0) * SEQ;   // Vt row = feature
    const bf16_t* Bl = Vtl + ((size_t)b * N_DIM + n0) * SEQ;

    v8f c00 = {}, c01 = {}, c10 = {}, c11 = {};
#pragma unroll 2
    for (int k = 0; k < SEQ; k += 32) {
        v16bf a0h = load_a_bf(Ph + k, SEQ, lane);
        v16bf a0l = load_a_bf(Pl + k, SEQ, lane);
        v16bf a1h = load_a_bf(Ph + (size_t)16 * SEQ + k, SEQ, lane);
        v16bf a1l = load_a_bf(Pl + (size_t)16 * SEQ + k, SEQ, lane);
        v16bf b0h = load_bt_bf(Bh + k, SEQ, lane);
        v16bf b0l = load_bt_bf(Bl + k, SEQ, lane);
        v16bf b1h = load_bt_bf(Bh + (size_t)16 * SEQ + k, SEQ, lane);
        v16bf b1l = load_bt_bf(Bl + (size_t)16 * SEQ + k, SEQ, lane);
        c00 = wmma3(a0h, a0l, b0h, b0l, c00);
        c01 = wmma3(a0h, a0l, b1h, b1l, c01);
        c10 = wmma3(a1h, a1l, b0h, b0l, c10);
        c11 = wmma3(a1h, a1l, b1h, b1l, c11);
    }

    float* C = O + ((size_t)b * SEQ + m0) * N_DIM + n0;
    store_c_f32(C, N_DIM, lane, c00);
    store_c_f32(C + 16, N_DIM, lane, c01);
    store_c_f32(C + (size_t)16 * N_DIM, N_DIM, lane, c10);
    store_c_f32(C + (size_t)16 * N_DIM + 16, N_DIM, lane, c11);
}

extern "C" void kernel_launch(void* const* d_in, const int* in_sizes, int n_in,
                              void* d_out, int out_size, void* d_ws, size_t ws_size,
                              hipStream_t stream) {
    const float* X = (const float*)d_in[0];   // [8,2048,768]
    const float* W = (const float*)d_in[1];   // [768,2304]
    float* out = (float*)d_out;               // [8,2048,768]

    // Workspace layout (bf16 hi/lo pairs); total ~342 MB.
    bf16_t* p   = (bf16_t*)d_ws;
    bf16_t* Xh  = p;  p += (size_t)ROWS * N_DIM;
    bf16_t* Xl  = p;  p += (size_t)ROWS * N_DIM;
    bf16_t* WTh = p;  p += (size_t)(3 * N_DIM) * N_DIM;
    bf16_t* WTl = p;  p += (size_t)(3 * N_DIM) * N_DIM;
    bf16_t* QKh = p;  p += (size_t)ROWS * QK_LD;
    bf16_t* QKl = p;  p += (size_t)ROWS * QK_LD;
    bf16_t* Vth = p;  p += (size_t)BATCH * N_DIM * SEQ;
    bf16_t* Vtl = p;  p += (size_t)BATCH * N_DIM * SEQ;
    bf16_t* Sh  = p;  p += (size_t)BATCH * SEQ * SEQ;
    bf16_t* Sl  = p;  p += (size_t)BATCH * SEQ * SEQ;

    // 0) one-time operand splits
    split_x_kernel <<<dim3(ROWS * N_DIM / 4 / 256), 256, 0, stream>>>(X, Xh, Xl);
    split_wT_kernel<<<dim3(3 * N_DIM * N_DIM / 256), 256, 0, stream>>>(W, WTh, WTl);
    // 1) QKV projection (v stored transposed)
    qkv_gemm_kernel<<<dim3(512, 9), 256, 0, stream>>>(Xh, Xl, WTh, WTl, QKh, QKl, Vth, Vtl);
    // 2) scaled scores
    scores_gemm_kernel<<<dim3(64, 8, BATCH), 256, 0, stream>>>(QKh, QKl, Sh, Sl);
    // 3) row softmax (in place)
    softmax_rows_kernel<<<dim3(ROWS), 256, 0, stream>>>(Sh, Sl);
    // 4) attention @ v
    pv_gemm_kernel<<<dim3(64, 3, BATCH), 256, 0, stream>>>(Sh, Sl, Vth, Vtl, out);
}
